// BinaryMLP_54202487276022
// MI455X (gfx1250) — 8-bit matrix operand order repaired and checked by reading, compiled, not run on hardware
//
#include <hip/hip_runtime.h>
#include <cstdint>

#define BN_EPS 1e-5f

typedef __attribute__((ext_vector_type(8))) int v8i;

__device__ __forceinline__ int8_t signof_f(float x) {
    return x > 0.f ? (int8_t)1 : (x < 0.f ? (int8_t)-1 : (int8_t)0);
}

__device__ __forceinline__ v8i load_a_frag(const int8_t* __restrict__ ap) {
    int4 a0 = *(const int4*)(ap);
    int4 a1 = *(const int4*)(ap + 32);
    v8i r = {a0.x, a0.y, a0.z, a0.w, a1.x, a1.y, a1.z, a1.w};
    return r;
}

__device__ __forceinline__ v8i load_b_frag(const int8_t* __restrict__ bp) {
    int4 b0 = *(const int4*)(bp);
    int4 b1 = *(const int4*)(bp + 32);
    v8i r = {b0.x, b0.y, b0.z, b0.w, b1.x, b1.y, b1.z, b1.w};
    return r;
}

__device__ __forceinline__ v8i bmma(v8i a, v8i b, v8i c) {
    return __builtin_amdgcn_wmma_i32_16x16x64_iu8(true, a, true, b, c, false, false);
}

__global__ void binarize_f32_kernel(const float* __restrict__ src,
                                    int8_t* __restrict__ dst,
                                    long long n_src, long long n_dst) {
    long long i = (long long)blockIdx.x * blockDim.x + threadIdx.x;
    if (i >= n_dst) return;
    dst[i] = (i < n_src) ? signof_f(src[i]) : (int8_t)1;
}

template <bool FINAL>
__global__ void bgemm_iu8_kernel(const int8_t* __restrict__ A,
                                 const int8_t* __restrict__ B,
                                 int* __restrict__ Y, float* __restrict__ Out,
                                 int M, int N, int K, int n_valid) {
    const int lane = threadIdx.x & 31;
    const int wave = threadIdx.x >> 5;
    const int half = lane >> 4;
    const int lm   = lane & 15;

    const int ntg_count = N >> 6;
    const int gwave = blockIdx.x * (blockDim.x >> 5) + wave;
    const int mt  = gwave / ntg_count;
    const int ntg = gwave - mt * ntg_count;
    if (mt >= (M >> 4)) return;

    const int8_t* __restrict__ ap = A + (size_t)(mt * 16 + lm) * K + half * 16;
    const int8_t* __restrict__ bp = B + (size_t)(ntg * 64 + lm) * K + half * 16;
    const size_t bstride = (size_t)16 * K;

    v8i acc0 = {}, acc1 = {}, acc2 = {}, acc3 = {};

    v8i a_c  = load_a_frag(ap);
    v8i b_c0 = load_b_frag(bp);
    v8i b_c1 = load_b_frag(bp + bstride);
    v8i b_c2 = load_b_frag(bp + 2 * bstride);
    v8i b_c3 = load_b_frag(bp + 3 * bstride);

    for (int kb = 64; kb < K; kb += 64) {
        v8i a_n  = load_a_frag(ap + kb);
        v8i b_n0 = load_b_frag(bp + kb);
        v8i b_n1 = load_b_frag(bp + bstride + kb);
        v8i b_n2 = load_b_frag(bp + 2 * bstride + kb);
        v8i b_n3 = load_b_frag(bp + 3 * bstride + kb);
        __builtin_prefetch((const void*)(ap + kb + 256), 0, 1);

        acc0 = bmma(a_c, b_c0, acc0);
        acc1 = bmma(a_c, b_c1, acc1);
        acc2 = bmma(a_c, b_c2, acc2);
        acc3 = bmma(a_c, b_c3, acc3);

        a_c = a_n;
        b_c0 = b_n0; b_c1 = b_n1; b_c2 = b_n2; b_c3 = b_n3;
    }

    acc0 = bmma(a_c, b_c0, acc0);
    acc1 = bmma(a_c, b_c1, acc1);
    acc2 = bmma(a_c, b_c2, acc2);
    acc3 = bmma(a_c, b_c3, acc3);

    const int row0 = mt * 16 + half * 8;
    const int col0 = ntg * 64 + lm;
#pragma unroll
    for (int r = 0; r < 8; ++r) {
        const int row = row0 + r;
        if (FINAL) {
            float* orow = Out + (size_t)row * n_valid;
            if (col0      < n_valid) orow[col0]      = (float)acc0[r];
            if (col0 + 16 < n_valid) orow[col0 + 16] = (float)acc1[r];
            if (col0 + 32 < n_valid) orow[col0 + 32] = (float)acc2[r];
            if (col0 + 48 < n_valid) orow[col0 + 48] = (float)acc3[r];
        } else {
            int* yrow = Y + (size_t)row * N + col0;
            yrow[0]  = acc0[r];
            yrow[16] = acc1[r];
            yrow[32] = acc2[r];
            yrow[48] = acc3[r];
        }
    }
}

__global__ void bn_colstats_kernel(const int* __restrict__ Y,
                                   const float* __restrict__ gamma,
                                   const float* __restrict__ beta,
                                   float* __restrict__ scale,
                                   float* __restrict__ bias,
                                   int Mrows, int N) {
    const int tid = threadIdx.x;
    const int c   = tid & 31;
    const int w   = tid >> 5;
    const int n   = blockIdx.x * 32 + c;

    double s = 0.0, q = 0.0;
    for (int r = w; r < Mrows; r += 8) {
        double v = (double)Y[(size_t)r * N + n];
        s += v;
        q += v * v;
    }

    __shared__ double sh_s[8][32];
    __shared__ double sh_q[8][32];
    sh_s[w][c] = s;
    sh_q[w][c] = q;
    __syncthreads();

    if (w == 0) {
#pragma unroll
        for (int i = 1; i < 8; ++i) {
            s += sh_s[i][c];
            q += sh_q[i][c];
        }
        double mean = s / (double)Mrows;
        double var  = q / (double)Mrows - mean * mean;
        float rr = rsqrtf((float)var + BN_EPS);
        float sc = gamma[n] * rr;
        scale[n] = sc;
        bias[n]  = beta[n] - sc * (float)mean;
    }
}

__global__ void bn_sign_kernel(const int* __restrict__ Y,
                               const float* __restrict__ scale,
                               const float* __restrict__ bias,
                               int8_t* __restrict__ dst,
                               long long total, int nmask) {
    long long i = (long long)blockIdx.x * blockDim.x + threadIdx.x;
    if (i >= total) return;
    int n = (int)i & nmask;
    float v = scale[n] * (float)Y[i] + bias[n];
    dst[i] = signof_f(v);
}

extern "C" void kernel_launch(void* const* d_in, const int* in_sizes, int n_in,
                              void* d_out, int out_size, void* d_ws, size_t ws_size,
                              hipStream_t stream) {
    const float* x    = (const float*)d_in[0];
    const float* W1   = (const float*)d_in[1];
    const float* g1   = (const float*)d_in[2];
    const float* b1   = (const float*)d_in[3];
    const float* W2   = (const float*)d_in[4];
    const float* g2   = (const float*)d_in[5];
    const float* b2   = (const float*)d_in[6];
    const float* W3   = (const float*)d_in[7];
    const float* g3   = (const float*)d_in[8];
    const float* b3   = (const float*)d_in[9];
    const float* Wout = (const float*)d_in[10];
    float* out = (float*)d_out;

    const int Bsz = 8192, D = 4096, H = 4096, C = 1000, Cpad = 1024;

    uint8_t* ws = (uint8_t*)d_ws;
    size_t off = 0;
    int8_t* Xb = (int8_t*)(ws + off); off += (size_t)Bsz * D;
    int8_t* Hb = (int8_t*)(ws + off); off += (size_t)Bsz * H;
    int8_t* Wb = (int8_t*)(ws + off); off += (size_t)H * H;
    int*    Yb = (int*)   (ws + off); off += (size_t)Bsz * H * 4;
    float* scale = (float*)(ws + off); off += (size_t)H * 4;
    float* bias  = (float*)(ws + off); off += (size_t)H * 4;
    (void)ws_size; (void)in_sizes; (void)n_in; (void)out_size;

    const long long elemsXH = (long long)Bsz * H;
    const int T = 256;

    auto bin = [&](const float* s, int8_t* d, long long nsrc, long long ndst) {
        binarize_f32_kernel<<<(unsigned)((ndst + T - 1) / T), T, 0, stream>>>(s, d, nsrc, ndst);
    };
    auto gemm = [&](const int8_t* A, const int8_t* Bm, int M, int N, int K) {
        int tiles = (M / 16) * (N / 64);
        bgemm_iu8_kernel<false><<<tiles / 8, T, 0, stream>>>(A, Bm, Yb, nullptr, M, N, K, N);
    };
    auto bnlayer = [&](const float* g, const float* b, int8_t* nxt) {
        bn_colstats_kernel<<<H / 32, T, 0, stream>>>(Yb, g, b, scale, bias, Bsz, H);
        bn_sign_kernel<<<(unsigned)((elemsXH + T - 1) / T), T, 0, stream>>>(
            Yb, scale, bias, nxt, elemsXH, H - 1);
    };

    bin(x, Xb, (long long)Bsz * D, (long long)Bsz * D);

    bin(W1, Wb, (long long)H * D, (long long)H * D);
    gemm(Xb, Wb, Bsz, H, D);
    bnlayer(g1, b1, Hb);

    bin(W2, Wb, (long long)H * H, (long long)H * H);
    gemm(Hb, Wb, Bsz, H, H);
    bnlayer(g2, b2, Xb);

    bin(W3, Wb, (long long)H * H, (long long)H * H);
    gemm(Xb, Wb, Bsz, H, H);
    bnlayer(g3, b3, Hb);

    bin(Wout, Wb, (long long)C * H, (long long)Cpad * H);
    {
        int tiles = (Bsz / 16) * (Cpad / 64);
        bgemm_iu8_kernel<true><<<tiles / 8, T, 0, stream>>>(
            Hb, Wb, nullptr, out, Bsz, Cpad, H, C);
    }
}
